// CustomGNN_66881230733874
// MI455X (gfx1250) — matrix operand order repaired and checked by reading, not compiled, not run on hardware
//
#include <hip/hip_runtime.h>

// ---------------------------------------------------------------------------
// CDNA5 (gfx1250) GAT implementation, round 3.
// - All matmuls via v_wmma_f32_16x16x32_bf16 (wave32 WMMA).
// - All WMMA operands load as contiguous 32-bit bf16 pairs (-> b128 merges):
//     * weights pre-transposed to [Nc][K] bf16,
//     * V written transposed per (b,h) by its projection GEMM,
//     * softmax writes unnormalized P once as bf16 into LDS (phase-3 A-frags
//       are straight ds loads, no per-iteration conversion).
// - GEMM: 16x64 tile per wave (A-frag reused over 4 B-frags).
// - Attention phase 3: one K-loop computes all D/64 column tiles per wave
//   (P-frag loaded once, multiple independent WMMAs per K-step).
// ---------------------------------------------------------------------------

typedef __attribute__((ext_vector_type(16))) __bf16 v16bf;
typedef __attribute__((ext_vector_type(8)))  float  v8f;

#define HEADS  8
#define NNODES 1024
#define BATCH  8

__device__ __forceinline__ unsigned short f32_to_bf16(float f) {
    unsigned u = __float_as_uint(f);
    unsigned r = 0x7FFFu + ((u >> 16) & 1u);   // round to nearest even
    return (unsigned short)((u + r) >> 16);
}

union FragAB { unsigned u[8]; v16bf v; };

// ---------------------------------------------------------------------------
// f32 -> bf16 conversion (plain)
// ---------------------------------------------------------------------------
__global__ void cvt_f32_bf16(const float* __restrict__ in,
                             unsigned short* __restrict__ out, int n) {
    int i = blockIdx.x * blockDim.x + threadIdx.x;
    if (i < n) out[i] = f32_to_bf16(in[i]);
}

// ---------------------------------------------------------------------------
// f32 [K][Nc] -> bf16 [Nc][K] transpose-convert (weights; one-time, tiny)
// ---------------------------------------------------------------------------
__global__ void cvt_f32_bf16_T(const float* __restrict__ in,
                               unsigned short* __restrict__ out,
                               int K, int Nc) {
    int i = blockIdx.x * blockDim.x + threadIdx.x;
    if (i >= K * Nc) return;
    int k = i / Nc, n = i - k * Nc;
    out[(size_t)n * K + k] = f32_to_bf16(in[i]);
}

// ---------------------------------------------------------------------------
// bf16 WMMA GEMM: C[M,Nc] = A[M,K] * W[K,Nc], with W supplied TRANSPOSED
// as Wt[Nc][K] bf16.  Each wave computes a 16x64 C tile (4 accumulators).
// OUTMODE: 0 = f32 row-major, 1 = bf16 row-major,
//          2 = bf16 transposed per head: out[((b*H+h)*HD + d)*NNODES + n]
// ---------------------------------------------------------------------------
template<int OUTMODE, int HD>
__global__ __launch_bounds__(128)
void gemm_wmma_bf16(const unsigned short* __restrict__ A,
                    const unsigned short* __restrict__ Wt,
                    void* __restrict__ Cout,
                    int M, int K, int Nc) {
    const int wave = threadIdx.x >> 5;
    const int lane = threadIdx.x & 31;
    const int tilesN = Nc >> 6;                       // 64-wide col tiles
    const int tiles  = (M >> 4) * tilesN;
    const int wid = blockIdx.x * (blockDim.x >> 5) + wave;
    if (wid >= tiles) return;                         // wave-uniform exit

    const int tm   = wid / tilesN;
    const int tn   = wid - tm * tilesN;
    const int half = (lane < 16) ? 0 : 1;
    const int l15  = lane & 15;
    const int am   = tm * 16 + l15;                   // A row for this lane

    v8f acc0 = {}, acc1 = {}, acc2 = {}, acc3 = {};
    const size_t arow = (size_t)am * K;
    const size_t brow0 = (size_t)(tn * 64 +  0 + l15) * K;
    const size_t brow1 = (size_t)(tn * 64 + 16 + l15) * K;
    const size_t brow2 = (size_t)(tn * 64 + 32 + l15) * K;
    const size_t brow3 = (size_t)(tn * 64 + 48 + l15) * K;

    for (int kc = 0; kc < K; kc += 32) {
        FragAB a, b0, b1, b2, b3;
        #pragma unroll
        for (int i = 0; i < 8; ++i) {
            int ka = kc + half * 16 + 2 * i;
            a.u[i] = *(const unsigned*)(A + arow + ka);
            int kb = kc + half * 16 + 2 * i;          // contiguous pair in Wt row
            b0.u[i] = *(const unsigned*)(Wt + brow0 + kb);
            b1.u[i] = *(const unsigned*)(Wt + brow1 + kb);
            b2.u[i] = *(const unsigned*)(Wt + brow2 + kb);
            b3.u[i] = *(const unsigned*)(Wt + brow3 + kb);
        }
        acc0 = __builtin_amdgcn_wmma_f32_16x16x32_bf16(false, a.v, false, b0.v, (short)0, acc0, false, false);
        acc1 = __builtin_amdgcn_wmma_f32_16x16x32_bf16(false, a.v, false, b1.v, (short)0, acc1, false, false);
        acc2 = __builtin_amdgcn_wmma_f32_16x16x32_bf16(false, a.v, false, b2.v, (short)0, acc2, false, false);
        acc3 = __builtin_amdgcn_wmma_f32_16x16x32_bf16(false, a.v, false, b3.v, (short)0, acc3, false, false);
    }

    #pragma unroll
    for (int j = 0; j < 4; ++j) {
        const v8f& acc = (j == 0) ? acc0 : (j == 1) ? acc1 : (j == 2) ? acc2 : acc3;
        const int coln = tn * 64 + j * 16 + l15;
        #pragma unroll
        for (int i = 0; i < 8; ++i) {
            const int row = tm * 16 + i + half * 8;
            if (OUTMODE == 0) {
                ((float*)Cout)[(size_t)row * Nc + coln] = acc[i];
            } else if (OUTMODE == 1) {
                ((unsigned short*)Cout)[(size_t)row * Nc + coln] = f32_to_bf16(acc[i]);
            } else {
                // transpose per head: row = b*NNODES+n ; coln = h*HD+d
                const int b = row >> 10, n = row & (NNODES - 1);
                const int h = coln / HD, d = coln - h * HD;
                ((unsigned short*)Cout)[((size_t)((b * HEADS + h) * HD + d)) * NNODES + n]
                    = f32_to_bf16(acc[i]);
            }
        }
    }
}

// ---------------------------------------------------------------------------
// Fused masked attention for one (batch, head, 16-row block).
// 4 waves / workgroup. LDS: f32 scores strip + bf16 P strip (dynamic).
//   phase1: S = Q K^T * scale + A * -1e9                 (WMMA)
//   phase2: rowwise softmax; write unnormalized exp as bf16 P strip
//   phase3: ctx = P @ Vt, /rowsum  (all col-tiles of a wave in one K-loop)
// ---------------------------------------------------------------------------
#define SCP (NNODES + 8)    // padded f32 LDS row stride
#define PSP (NNODES + 16)   // padded bf16 LDS row stride

template<int D>
__global__ __launch_bounds__(128)
void gat_attention(const unsigned short* __restrict__ Q,
                   const unsigned short* __restrict__ Kg,
                   const unsigned short* __restrict__ Vt,  // [(b*H+h)*D + d][n]
                   const float* __restrict__ Adj,
                   unsigned short* __restrict__ ctx) {
    constexpr int STRIDE = HEADS * D;      // Q/K/ctx row stride (elements)
    constexpr int NT = NNODES / 16;        // 64 key tiles
    constexpr int CPW = D / 64;            // col tiles per wave in phase 3

    extern __shared__ char smem[];
    float*          sc     = (float*)smem;            // [16][SCP] f32 scores
    float*          rowsum = sc + 16 * SCP;           // [16]
    unsigned short* Pb     = (unsigned short*)(rowsum + 16); // [16][PSP] bf16 P

    int idx = blockIdx.x;
    const int rblk = idx & (NT - 1); idx >>= 6;
    const int h    = idx & (HEADS - 1); idx >>= 3;
    const int b    = idx;
    const int row0 = rblk * 16;

    const int wave = threadIdx.x >> 5;
    const int lane = threadIdx.x & 31;
    const int half = (lane < 16) ? 0 : 1;
    const int l15  = lane & 15;
    const float scale = (D == 128) ? 0.08838834764831845f : 0.125f;

    const size_t qrow = ((size_t)(b * NNODES + row0 + l15)) * STRIDE + h * D;

    // ---- phase 1: scores -------------------------------------------------
    for (int nt = wave; nt < NT; nt += 4) {
        const int n0 = nt * 16;
        const size_t krow = ((size_t)(b * NNODES + n0 + l15)) * STRIDE + h * D;
        if (nt + 4 < NT)   // prefetch next key tile for this wave
            __builtin_prefetch(Kg + krow + (size_t)64 * STRIDE, 0, 1);
        v8f acc = {};
        #pragma unroll
        for (int kc = 0; kc < D; kc += 32) {
            FragAB a, bb;
            #pragma unroll
            for (int i = 0; i < 8; ++i) {
                int ka = kc + half * 16 + 2 * i;
                a.u[i]  = *(const unsigned*)(Q  + qrow + ka);
                int kb = kc + half * 16 + 2 * i;      // K^T: contiguous in K row
                bb.u[i] = *(const unsigned*)(Kg + krow + kb);
            }
            acc = __builtin_amdgcn_wmma_f32_16x16x32_bf16(
                      false, a.v, false, bb.v, (short)0, acc, false, false);
        }
        const int col = n0 + l15;
        #pragma unroll
        for (int i = 0; i < 8; ++i) {
            int r = i + half * 8;
            float ae = Adj[((size_t)(b * NNODES + row0 + r)) * NNODES + col];
            sc[r * SCP + col] = acc[i] * scale - 1.0e9f * ae;
        }
    }
    __syncthreads();

    // ---- phase 2: softmax; emit unnormalized exp as bf16 into Pb ---------
    {
        const int r = threadIdx.x >> 3;
        const int s = threadIdx.x & 7;
        float mx = -3.4e38f;
        for (int c = s; c < NNODES; c += 8) mx = fmaxf(mx, sc[r * SCP + c]);
        #pragma unroll
        for (int off = 1; off < 8; off <<= 1) mx = fmaxf(mx, __shfl_xor(mx, off, 8));
        float sum = 0.f;
        for (int c = s; c < NNODES; c += 8) {
            float e = __expf(sc[r * SCP + c] - mx);
            Pb[r * PSP + c] = f32_to_bf16(e);   // P in [0,1]: bf16-safe
            sum += e;
        }
        #pragma unroll
        for (int off = 1; off < 8; off <<= 1) sum += __shfl_xor(sum, off, 8);
        if (s == 0) rowsum[r] = sum;
    }
    __syncthreads();

    // ---- phase 3: ctx = P @ Vt (all CPW col tiles in one K-loop) ---------
    {
        int col[CPW];
        size_t vtrow[CPW];
        v8f acc[CPW];
        #pragma unroll
        for (int j = 0; j < CPW; ++j) {
            col[j]   = (wave + 4 * j) * 16 + l15;     // column (d) within head
            vtrow[j] = ((size_t)((b * HEADS + h) * D + col[j])) * NNODES;
            acc[j]   = (v8f){};
        }
        const unsigned short* prow = Pb + l15 * PSP;

        for (int kc = 0; kc < NNODES; kc += 32) {
            FragAB a;
            #pragma unroll
            for (int i = 0; i < 8; ++i) {
                int ka = kc + half * 16 + 2 * i;
                a.u[i] = *(const unsigned*)(prow + ka);   // ds_load (merges)
            }
            #pragma unroll
            for (int j = 0; j < CPW; ++j) {
                if (kc + 128 < NNODES)
                    __builtin_prefetch(Vt + vtrow[j] + kc + 128, 0, 1);
                FragAB bb;
                #pragma unroll
                for (int i = 0; i < 8; ++i) {
                    int kb = kc + half * 16 + 2 * i;      // contiguous in Vt row
                    bb.u[i] = *(const unsigned*)(Vt + vtrow[j] + kb);
                }
                acc[j] = __builtin_amdgcn_wmma_f32_16x16x32_bf16(
                             false, a.v, false, bb.v, (short)0, acc[j], false, false);
            }
        }
        #pragma unroll
        for (int j = 0; j < CPW; ++j) {
            #pragma unroll
            for (int i = 0; i < 8; ++i) {
                int r = i + half * 8;
                float v = acc[j][i] / rowsum[r];   // fold softmax normalization
                ctx[((size_t)(b * NNODES + row0 + r)) * STRIDE + h * D + col[j]]
                    = f32_to_bf16(v);
            }
        }
    }
}

// ---------------------------------------------------------------------------
// mean over nodes: [B,N,64] -> [B,64]
// ---------------------------------------------------------------------------
__global__ void mean_nodes(const float* __restrict__ hsrc,
                           float* __restrict__ m) {
    int b = blockIdx.x, j = threadIdx.x;   // 64 threads
    float s = 0.f;
    for (int n = 0; n < NNODES; ++n)
        s += hsrc[((size_t)(b * NNODES + n)) * 64 + j];
    m[b * 64 + j] = s * (1.0f / NNODES);
}

// ---------------------------------------------------------------------------
// tiny MLP head: 64 -> 32 -> 16 -> 1 (per batch)
// ---------------------------------------------------------------------------
__global__ void mlp_head(const float* __restrict__ m,
                         const float* __restrict__ W1, const float* __restrict__ b1,
                         const float* __restrict__ W2, const float* __restrict__ b2,
                         const float* __restrict__ W3, const float* __restrict__ b3,
                         float* __restrict__ out) {
    int b = threadIdx.x;
    if (b >= BATCH) return;
    float t1[32];
    for (int j = 0; j < 32; ++j) {
        float s = b1[j];
        for (int i = 0; i < 64; ++i) s += m[b * 64 + i] * W1[i * 32 + j];
        t1[j] = fmaxf(s, 0.f);
    }
    float t2[16];
    for (int j = 0; j < 16; ++j) {
        float s = b2[j];
        for (int i = 0; i < 32; ++i) s += t1[i] * W2[i * 16 + j];
        t2[j] = fmaxf(s, 0.f);
    }
    float s = b3[0];
    for (int i = 0; i < 16; ++i) s += t2[i] * W3[i];
    out[b] = s;
}

// ---------------------------------------------------------------------------
extern "C" void kernel_launch(void* const* d_in, const int* in_sizes, int n_in,
                              void* d_out, int out_size, void* d_ws, size_t ws_size,
                              hipStream_t stream) {
    const float* X   = (const float*)d_in[0];
    const float* A   = (const float*)d_in[1];
    const float* Wq1 = (const float*)d_in[2];
    const float* Wk1 = (const float*)d_in[3];
    const float* Wv1 = (const float*)d_in[4];
    const float* Wo1 = (const float*)d_in[5];
    const float* Wq2 = (const float*)d_in[6];
    const float* Wk2 = (const float*)d_in[7];
    const float* Wv2 = (const float*)d_in[8];
    const float* Wo2 = (const float*)d_in[9];
    const float* W1  = (const float*)d_in[10];
    const float* b1  = (const float*)d_in[11];
    const float* W2  = (const float*)d_in[12];
    const float* b2  = (const float*)d_in[13];
    const float* W3  = (const float*)d_in[14];
    const float* b3  = (const float*)d_in[15];

    // ---- workspace carve-out (bump allocator, 256B aligned) -------------
    char*  ws  = (char*)d_ws;
    size_t off = 0;
    auto alloc = [&](size_t bytes) -> char* {
        char* p = ws + off;
        off = (off + bytes + 255) & ~(size_t)255;
        return p;
    };
    const int M = BATCH * NNODES;                        // 8192

    unsigned short* Xb   = (unsigned short*)alloc((size_t)M * 64 * 2);
    unsigned short* Wq1t = (unsigned short*)alloc(64 * 1024 * 2);   // [Nc][K]
    unsigned short* Wk1t = (unsigned short*)alloc(64 * 1024 * 2);
    unsigned short* Wv1t = (unsigned short*)alloc(64 * 1024 * 2);
    unsigned short* Wo1t = (unsigned short*)alloc(1024 * 128 * 2);
    unsigned short* Wq2t = (unsigned short*)alloc(128 * 512 * 2);
    unsigned short* Wk2t = (unsigned short*)alloc(128 * 512 * 2);
    unsigned short* Wv2t = (unsigned short*)alloc(128 * 512 * 2);
    unsigned short* Wo2t = (unsigned short*)alloc(512 * 64 * 2);
    unsigned short* Qb   = (unsigned short*)alloc((size_t)M * 1024 * 2); // L2 reuses
    unsigned short* Kb   = (unsigned short*)alloc((size_t)M * 1024 * 2);
    unsigned short* Vtb  = (unsigned short*)alloc((size_t)M * 1024 * 2); // V transposed
    unsigned short* CTX  = (unsigned short*)alloc((size_t)M * 1024 * 2); // L2 reuses
    unsigned short* H1b  = (unsigned short*)alloc((size_t)M * 128 * 2);
    float*          H2f  = (float*)alloc((size_t)M * 64 * 4);
    float*          MEAN = (float*)alloc(BATCH * 64 * 4);
    (void)ws_size; (void)in_sizes; (void)n_in; (void)out_size;

    auto cdiv = [](int a, int b) { return (a + b - 1) / b; };

    // ---- bf16 conversions (weights transposed) ---------------------------
    cvt_f32_bf16<<<cdiv(M * 64, 256), 256, 0, stream>>>(X, Xb, M * 64);
    cvt_f32_bf16_T<<<cdiv(64 * 1024, 256), 256, 0, stream>>>(Wq1, Wq1t, 64, 1024);
    cvt_f32_bf16_T<<<cdiv(64 * 1024, 256), 256, 0, stream>>>(Wk1, Wk1t, 64, 1024);
    cvt_f32_bf16_T<<<cdiv(64 * 1024, 256), 256, 0, stream>>>(Wv1, Wv1t, 64, 1024);
    cvt_f32_bf16_T<<<cdiv(1024 * 128, 256), 256, 0, stream>>>(Wo1, Wo1t, 1024, 128);
    cvt_f32_bf16_T<<<cdiv(128 * 512, 256), 256, 0, stream>>>(Wq2, Wq2t, 128, 512);
    cvt_f32_bf16_T<<<cdiv(128 * 512, 256), 256, 0, stream>>>(Wk2, Wk2t, 128, 512);
    cvt_f32_bf16_T<<<cdiv(128 * 512, 256), 256, 0, stream>>>(Wv2, Wv2t, 128, 512);
    cvt_f32_bf16_T<<<cdiv(512 * 64, 256), 256, 0, stream>>>(Wo2, Wo2t, 512, 64);

    const size_t shmem = (size_t)16 * SCP * 4 + 16 * 4 + (size_t)16 * PSP * 2;
    (void)hipFuncSetAttribute((const void*)&gat_attention<128>,
                              hipFuncAttributeMaxDynamicSharedMemorySize, (int)shmem);
    (void)hipFuncSetAttribute((const void*)&gat_attention<64>,
                              hipFuncAttributeMaxDynamicSharedMemorySize, (int)shmem);

    // ---- layer 1 ---------------------------------------------------------
    {
        int tiles = (M / 16) * (1024 / 64);
        gemm_wmma_bf16<1, 0><<<cdiv(tiles, 4), 128, 0, stream>>>(Xb, Wq1t, Qb, M, 64, 1024);
        gemm_wmma_bf16<1, 0><<<cdiv(tiles, 4), 128, 0, stream>>>(Xb, Wk1t, Kb, M, 64, 1024);
        gemm_wmma_bf16<2, 128><<<cdiv(tiles, 4), 128, 0, stream>>>(Xb, Wv1t, Vtb, M, 64, 1024);

        gat_attention<128><<<BATCH * HEADS * (NNODES / 16), 128, shmem, stream>>>(
            Qb, Kb, Vtb, A, CTX);

        int tilesO = (M / 16) * (128 / 64);
        gemm_wmma_bf16<1, 0><<<cdiv(tilesO, 4), 128, 0, stream>>>(CTX, Wo1t, H1b, M, 1024, 128);
    }

    // ---- layer 2 (reuses Q/K/Vt/CTX buffers) -----------------------------
    {
        int tiles = (M / 16) * (512 / 64);
        gemm_wmma_bf16<1, 0><<<cdiv(tiles, 4), 128, 0, stream>>>(H1b, Wq2t, Qb, M, 128, 512);
        gemm_wmma_bf16<1, 0><<<cdiv(tiles, 4), 128, 0, stream>>>(H1b, Wk2t, Kb, M, 128, 512);
        gemm_wmma_bf16<2, 64><<<cdiv(tiles, 4), 128, 0, stream>>>(H1b, Wv2t, Vtb, M, 128, 512);

        gat_attention<64><<<BATCH * HEADS * (NNODES / 16), 128, shmem, stream>>>(
            Qb, Kb, Vtb, A, CTX);

        int tilesO = (M / 16) * (64 / 64);
        gemm_wmma_bf16<0, 0><<<cdiv(tilesO, 4), 128, 0, stream>>>(CTX, Wo2t, H2f, M, 512, 64);
    }

    // ---- readout ---------------------------------------------------------
    mean_nodes<<<BATCH, 64, 0, stream>>>(H2f, MEAN);
    mlp_head<<<1, 32, 0, stream>>>(MEAN, W1, b1, W2, b2, W3, b3, (float*)d_out);
}
